// Model_9964324126949
// MI455X (gfx1250) — matrix operand order repaired and checked by reading, not compiled, not run on hardware
//
#include <hip/hip_runtime.h>
#include <math.h>

// ---------------------------------------------------------------------------
// Types for CDNA5 WMMA (wave32): v_wmma_f32_16x16x32_f16
// ---------------------------------------------------------------------------
typedef __attribute__((ext_vector_type(16))) _Float16 v16h;
typedef __attribute__((ext_vector_type(2)))  _Float16 v2h;
typedef __attribute__((ext_vector_type(8)))  float    v8f;

#define BATCH 32768
#define JOINTS 57

// ---------------------------------------------------------------------------
// PARENTS table: reference builds it from numpy default_rng(7); that stream
// cannot be evaluated here, so we generate a structurally identical tree
// (parent[j] < j, parent[0] = -1) with a deterministic constexpr LCG.
// ---------------------------------------------------------------------------
struct ParentTab { int p[JOINTS]; };
static constexpr ParentTab make_parents() {
    ParentTab t{};
    t.p[0] = -1;
    unsigned s = 7u;
    for (int j = 1; j < JOINTS; ++j) {
        s = s * 1664525u + 1013904223u;
        t.p[j] = (int)(s % (unsigned)j);
    }
    return t;
}
__constant__ ParentTab PAR = make_parents();

static __device__ inline v2h pack_f16(float a, float b) {
    v2h p;
    p.x = (_Float16)a;
    p.y = (_Float16)b;
    return p;
}

// ---------------------------------------------------------------------------
// Fused GEMM + bias + (leaky)ReLU using WMMA f16 -> f32.
//   C[M,N] = act(A[M,K] @ W[K,N] + b[N])
// Workgroup tile: 128(M) x 128(N), K-step 32. 8 waves, each 32x64.
// Double-buffered LDS staging; branch-free clamped-index loads (CK/CN);
// f16 epilogue transposes C through LDS for coalesced b128 stores.
// ---------------------------------------------------------------------------
union GemmSmem {
    _Float16 stage[2][2][128][40];   // [A/B][buf][row][k], 16B row pad
    _Float16 ctile[128][136];        // C output tile (epilogue reuse)
};

template <int A_F32, int RELU, int OUT_F32, int CK, int CN>
__global__ __launch_bounds__(256) void gemm_kernel(
    const void* __restrict__ Aptr,     // f32 (A_F32) or f16 activations
    const float* __restrict__ W,       // f32 weights [K][N]
    const float* __restrict__ bias,    // f32 [N]
    void* __restrict__ Out,            // f16 activations or f32
    int M, int N, int K, int Astride, int Nout)
{
    __shared__ GemmSmem sm;

    const int tid   = threadIdx.x;
    const int mBase = blockIdx.y * 128;
    const int nBase = blockIdx.x * 128;

    const int wave = tid >> 5;
    const int lane = tid & 31;
    const int wm   = wave & 3;         // 4 row groups of 32
    const int wn   = wave >> 2;        // 2 col groups of 64
    const int lid  = lane & 15;
    const int lh   = lane >> 4;

    // staging decomposition (256 threads):
    //  A: (row = tid>>1, seg = tid&1) -> 16 f16 elems (32B)
    //  B: (kpair = (tid&15)*2, n0 = (tid>>4)*8) -> 2 K-rows x 8 N
    const int arow = tid >> 1;
    const int aseg = tid & 1;
    const int bkp  = (tid & 15) * 2;
    const int bn0  = (tid >> 4) * 8;

    v8f acc[2][4];
    const v8f vzero = {0.f, 0.f, 0.f, 0.f, 0.f, 0.f, 0.f, 0.f};
#pragma unroll
    for (int mt = 0; mt < 2; ++mt)
#pragma unroll
        for (int nt = 0; nt < 4; ++nt)
            acc[mt][nt] = vzero;

    union AReg { uint4 q[2]; _Float16 h[16]; };

    // ---- branch-free tile load: global -> registers ----
    auto load_tile = [&](int k0, AReg& ar, float* br) {
        if (A_F32) {
            const float* A = (const float*)Aptr;
            const size_t rowoff = (size_t)(mBase + arow) * (size_t)Astride;
            float av[16];
#pragma unroll
            for (int i = 0; i < 16; ++i) {
                int kk = k0 + aseg * 16 + i;
                int kc = CK ? (kk < K ? kk : K - 1) : kk;
                float v = A[rowoff + kc];           // always in-bounds
                if (CK && kk >= K) v = 0.0f;        // value select, no branch
                av[i] = v;
            }
#pragma unroll
            for (int i = 0; i < 16; ++i)
                ar.h[i] = (_Float16)av[i];
        } else {
            const _Float16* A = (const _Float16*)Aptr;
            const uint4* sq = (const uint4*)(A +
                (size_t)(mBase + arow) * (size_t)Astride + k0 + aseg * 16);
            ar.q[0] = sq[0];
            ar.q[1] = sq[1];
        }

        if (!CK && !CN) {
            const float* r0 = W + (size_t)(k0 + bkp) * (size_t)N + nBase + bn0;
            const float* r1 = r0 + N;
            float4 x0 = ((const float4*)r0)[0];
            float4 x1 = ((const float4*)r0)[1];
            float4 y0 = ((const float4*)r1)[0];
            float4 y1 = ((const float4*)r1)[1];
            br[0] = x0.x; br[1] = x0.y; br[2] = x0.z; br[3] = x0.w;
            br[4] = x1.x; br[5] = x1.y; br[6] = x1.z; br[7] = x1.w;
            br[8]  = y0.x; br[9]  = y0.y; br[10] = y0.z; br[11] = y0.w;
            br[12] = y1.x; br[13] = y1.y; br[14] = y1.z; br[15] = y1.w;
        } else {
            int k1 = k0 + bkp, k2 = k0 + bkp + 1;
            int k1c = CK ? (k1 < K ? k1 : K - 1) : k1;
            int k2c = CK ? (k2 < K ? k2 : K - 1) : k2;
            const float* r0 = W + (size_t)k1c * (size_t)N;
            const float* r1 = W + (size_t)k2c * (size_t)N;
#pragma unroll
            for (int i = 0; i < 8; ++i) {
                int nn = nBase + bn0 + i;
                int nc = CN ? (nn < N ? nn : N - 1) : nn;
                float v0 = r0[nc];
                float v1 = r1[nc];
                if (CK && k1 >= K) v0 = 0.0f;
                if (CK && k2 >= K) v1 = 0.0f;
                if (CN && nn >= N) { v0 = 0.0f; v1 = 0.0f; }
                br[i]     = v0;
                br[8 + i] = v1;
            }
        }
    };

    // ---- registers -> LDS (packed f16 stores) ----
    auto store_tile = [&](int buf, const AReg& ar, const float* br) {
        uint4* dq = (uint4*)&sm.stage[0][buf][arow][aseg * 16];
        dq[0] = ar.q[0];
        dq[1] = ar.q[1];
#pragma unroll
        for (int i = 0; i < 8; ++i) {
            v2h p = pack_f16(br[i], br[8 + i]);
            *(v2h*)&sm.stage[1][buf][bn0 + i][bkp] = p;  // one ds_store_b32
        }
    };

    const int nk = (K + 31) / 32;
    AReg areg;
    float breg[16];
    load_tile(0, areg, breg);
    store_tile(0, areg, breg);

    int cur = 0;
    for (int kt = 0; kt < nk; ++kt) {
        const bool pf = (kt + 1 < nk);
        AReg anext;
        float bnext[16];
        if (pf) {
            if (!CK && !CN)
                __builtin_prefetch(W + (size_t)(kt * 32 + 64 + bkp) * N +
                                       nBase + bn0, 0, 1);
            load_tile((kt + 1) * 32, anext, bnext);
        }

        __syncthreads();

        // ---- build fragments (two b128 loads each) + 8 WMMA ----
        union FragU { v16h v; uint4 q[2]; };
        v16h a[2], b[4];
#pragma unroll
        for (int mt = 0; mt < 2; ++mt) {
            const int r = wm * 32 + mt * 16 + lid;
            FragU u;
            u.q[0] = *(const uint4*)&sm.stage[0][cur][r][8 * lh];
            u.q[1] = *(const uint4*)&sm.stage[0][cur][r][16 + 8 * lh];
            a[mt] = u.v;
        }
#pragma unroll
        for (int nt = 0; nt < 4; ++nt) {
            const int c = wn * 64 + nt * 16 + lid;
            FragU u;
            u.q[0] = *(const uint4*)&sm.stage[1][cur][c][8 * lh];
            u.q[1] = *(const uint4*)&sm.stage[1][cur][c][16 + 8 * lh];
            b[nt] = u.v;
        }
#pragma unroll
        for (int mt = 0; mt < 2; ++mt)
#pragma unroll
            for (int nt = 0; nt < 4; ++nt)
                acc[mt][nt] = __builtin_amdgcn_wmma_f32_16x16x32_f16(
                    false, a[mt], false, b[nt], (short)0, acc[mt][nt],
                    false, false);

        if (pf) store_tile(cur ^ 1, anext, bnext);
        cur ^= 1;
    }

    // ---- epilogue: bias + leaky relu ----
    if (!OUT_F32) {
        // f16 path (N==1024, no guards): transpose C tile through LDS,
        // then fully coalesced 128B global stores.
        __syncthreads();   // staging buffers are dead; reuse as ctile
#pragma unroll
        for (int nt = 0; nt < 4; ++nt) {
            const int cc = wn * 64 + nt * 16 + lid;
            const float bv = bias[nBase + cc];
#pragma unroll
            for (int mt = 0; mt < 2; ++mt) {
#pragma unroll
                for (int r = 0; r < 8; ++r) {
                    float v = acc[mt][nt][r] + bv;
                    if (RELU) v = (v >= 0.0f) ? v : 0.01f * v;
                    sm.ctile[wm * 32 + mt * 16 + lh * 8 + r][cc] =
                        (_Float16)v;
                }
            }
        }
        __syncthreads();
        const int orow = tid >> 1;
        const int oseg = tid & 1;
        _Float16* dst = (_Float16*)Out +
            (size_t)(mBase + orow) * (size_t)Nout + nBase + oseg * 64;
        const uint4* src = (const uint4*)&sm.ctile[orow][oseg * 64];
#pragma unroll
        for (int i = 0; i < 8; ++i)
            ((uint4*)dst)[i] = src[i];
    } else {
        // f32 path (layer 4, N=228): guarded scalar stores
#pragma unroll
        for (int nt = 0; nt < 4; ++nt) {
            const int n = nBase + wn * 64 + nt * 16 + lid;
            const int nc = CN ? (n < N ? n : N - 1) : n;
            const float bv = bias[nc];
#pragma unroll
            for (int mt = 0; mt < 2; ++mt) {
#pragma unroll
                for (int r = 0; r < 8; ++r) {
                    const int m = mBase + wm * 32 + mt * 16 + lh * 8 + r;
                    float v = acc[mt][nt][r] + bv;
                    if (RELU) v = (v >= 0.0f) ? v : 0.01f * v;
                    if (!CN || n < N)
                        ((float*)Out)[(size_t)m * (size_t)Nout + n] = v;
                }
            }
        }
    }
}

// ---------------------------------------------------------------------------
// Quaternion -> 4x4 rotation matrix (rot output of the reference)
// ---------------------------------------------------------------------------
__global__ __launch_bounds__(256) void quat_kernel(
    const float* __restrict__ q, float* __restrict__ rot, int n)
{
    int i = blockIdx.x * blockDim.x + threadIdx.x;
    if (i >= n) return;
    float4 qv = ((const float4*)q)[i];
    float qw = qv.x, qx = qv.y, qy = qv.z, qz = qv.w;
    float nrm = sqrtf(qw * qw + qx * qx + qy * qy + qz * qz);
    nrm = fmaxf(nrm, 1e-8f);
    float inv = 1.0f / nrm;
    qw *= inv; qx *= inv; qy *= inv; qz *= inv;

    float xx = qx * qx, yy = qy * qy, zz = qz * qz;
    float xy = qx * qy, xz = qx * qz, yz = qy * qz;
    float xw = qx * qw, yw = qy * qw, zw = qz * qw;

    float4* o = (float4*)(rot + (size_t)i * 16);
    o[0] = make_float4(1.f - 2.f * (yy + zz), 2.f * (xy - zw), 2.f * (xz + yw), 0.f);
    o[1] = make_float4(2.f * (xy + zw), 1.f - 2.f * (xx + zz), 2.f * (yz - xw), 0.f);
    o[2] = make_float4(2.f * (xz - yw), 2.f * (yz + xw), 1.f - 2.f * (xx + yy), 0.f);
    o[3] = make_float4(0.f, 0.f, 0.f, 1.f);
}

// ---------------------------------------------------------------------------
// Forward kinematics: per-row chain over 57 joints; G kept as 3x4 affine
// (column-major, 12 floats) in a private array. pose = G[:, :3, 3].
// ---------------------------------------------------------------------------
__global__ __launch_bounds__(64) void fk_kernel(
    const float* __restrict__ rot, const float* __restrict__ joff,
    float* __restrict__ pose, int B)
{
    int row = blockIdx.x * 64 + threadIdx.x;
    if (row >= B) return;

    float G[JOINTS][12];

    for (int j = 0; j < JOINTS; ++j) {
        const float* R = rot + ((size_t)row * JOINTS + j) * 16;
        float r00 = R[0], r01 = R[1], r02 = R[2];
        float r10 = R[4], r11 = R[5], r12 = R[6];
        float r20 = R[8], r21 = R[9], r22 = R[10];
        float tx = joff[j * 3 + 0], ty = joff[j * 3 + 1], tz = joff[j * 3 + 2];

        float* g = G[j];
        int p = PAR.p[j];
        if (p < 0) {
            g[0] = r00; g[1] = r10; g[2] = r20;
            g[3] = r01; g[4] = r11; g[5] = r21;
            g[6] = r02; g[7] = r12; g[8] = r22;
            g[9] = tx;  g[10] = ty; g[11] = tz;
        } else {
            const float* gp = G[p];
            float a00 = gp[0], a10 = gp[1], a20 = gp[2];
            float a01 = gp[3], a11 = gp[4], a21 = gp[5];
            float a02 = gp[6], a12 = gp[7], a22 = gp[8];
            float a03 = gp[9], a13 = gp[10], a23 = gp[11];
            g[0] = a00 * r00 + a01 * r10 + a02 * r20;
            g[1] = a10 * r00 + a11 * r10 + a12 * r20;
            g[2] = a20 * r00 + a21 * r10 + a22 * r20;
            g[3] = a00 * r01 + a01 * r11 + a02 * r21;
            g[4] = a10 * r01 + a11 * r11 + a12 * r21;
            g[5] = a20 * r01 + a21 * r11 + a22 * r21;
            g[6] = a00 * r02 + a01 * r12 + a02 * r22;
            g[7] = a10 * r02 + a11 * r12 + a12 * r22;
            g[8] = a20 * r02 + a21 * r12 + a22 * r22;
            g[9]  = a00 * tx + a01 * ty + a02 * tz + a03;
            g[10] = a10 * tx + a11 * ty + a12 * tz + a13;
            g[11] = a20 * tx + a21 * ty + a22 * tz + a23;
        }
        float* po = pose + ((size_t)row * JOINTS + j) * 3;
        po[0] = g[9];
        po[1] = g[10];
        po[2] = g[11];
    }
}

// ---------------------------------------------------------------------------
// Host launcher
// ---------------------------------------------------------------------------
extern "C" void kernel_launch(void* const* d_in, const int* in_sizes, int n_in,
                              void* d_out, int out_size, void* d_ws,
                              size_t ws_size, hipStream_t stream)
{
    (void)in_sizes; (void)n_in; (void)out_size; (void)ws_size;

    const float* in_seq = (const float*)d_in[0];   // (B, 57, 3)
    const float* w1 = (const float*)d_in[1];       // (171, 1024)
    const float* b1 = (const float*)d_in[2];
    const float* w2 = (const float*)d_in[3];       // (1024, 1024)
    const float* b2 = (const float*)d_in[4];
    const float* w3 = (const float*)d_in[5];       // (1024, 1024)
    const float* b3 = (const float*)d_in[6];
    const float* w4 = (const float*)d_in[7];       // (1024, 228)
    const float* b4 = (const float*)d_in[8];
    const float* joff = (const float*)d_in[9];     // (57, 3)

    const int B = BATCH;
    const int Din = JOINTS * 3;                    // 171
    const int Dq = 4 * JOINTS;                     // 228

    // workspace layout
    char* ws = (char*)d_ws;
    _Float16* hA = (_Float16*)ws;                               // B*1024 f16
    _Float16* hB = (_Float16*)(ws + (size_t)B * 1024 * 2);      // B*1024 f16
    float* qbuf  = (float*)(ws + (size_t)B * 1024 * 4);         // B*228 f32

    float* pose = (float*)d_out;                   // B*57*3
    float* rot  = pose + (size_t)B * JOINTS * 3;   // B*57*16

    dim3 blk(256);
    dim3 grid1024(1024 / 128, B / 128);
    dim3 grid228((Dq + 127) / 128, B / 128);

    // layer 1: f32 in, K=171 (CK), lrelu, f16 out
    gemm_kernel<1, 1, 0, 1, 0><<<grid1024, blk, 0, stream>>>(
        (const void*)in_seq, w1, b1, (void*)hA, B, 1024, Din, Din, 1024);
    // layer 2: clean path, no bounds checks
    gemm_kernel<0, 1, 0, 0, 0><<<grid1024, blk, 0, stream>>>(
        (const void*)hA, w2, b2, (void*)hB, B, 1024, 1024, 1024, 1024);
    // layer 3
    gemm_kernel<0, 1, 0, 0, 0><<<grid1024, blk, 0, stream>>>(
        (const void*)hB, w3, b3, (void*)hA, B, 1024, 1024, 1024, 1024);
    // layer 4: no activation, N=228 (CN), f32 quaternion out
    gemm_kernel<0, 0, 1, 0, 1><<<grid228, blk, 0, stream>>>(
        (const void*)hA, w4, b4, (void*)qbuf, B, Dq, 1024, 1024, Dq);

    // quaternion -> rot (4x4) output
    int nq = B * JOINTS;
    quat_kernel<<<(nq + 255) / 256, 256, 0, stream>>>(qbuf, rot, nq);

    // forward kinematics -> pose output
    fk_kernel<<<B / 64, 64, 0, stream>>>(rot, joff, pose, B);
}
